// BestQuantumCNN_66211215835680
// MI455X (gfx1250) — matrix operand order repaired and checked by reading, not compiled, not run on hardware
//
#include <hip/hip_runtime.h>
#include <hip/hip_bf16.h>
#include <math.h>

// ---------------------------------------------------------------------------
// CDNA5 (gfx1250) hybrid quantum-CNN forward pass.
// wave32 WMMA f32_16x16x32_f16 for all conv (implicit GEMM) and linear layers.
// Conv weight panels staged to LDS via the Tensor Data Mover when available.
// ---------------------------------------------------------------------------

typedef __attribute__((ext_vector_type(16))) _Float16 v16h;
typedef __attribute__((ext_vector_type(8)))  float    v8f;
typedef __attribute__((ext_vector_type(4)))  float    f32x4;
typedef __attribute__((ext_vector_type(4)))  unsigned int u32x4;
typedef __attribute__((ext_vector_type(4)))  int      i32x4;
typedef __attribute__((ext_vector_type(8)))  int      i32x8;

#define BATCH 64

// Tensor Data Mover availability (arity differs across toolchains).
#if defined(__has_builtin)
#if __has_builtin(__builtin_amdgcn_tensor_load_to_lds) && \
    __has_builtin(__builtin_amdgcn_s_wait_tensorcnt)
#define HAVE_TDM 1
#endif
#endif
#ifndef HAVE_TDM
#define HAVE_TDM 0
#endif

// ---- fragment loaders -----------------------------------------------------
// A-matrix 16x32 f16 layout (ISA 7.12.2): lanes 0-15 hold row M=lane,
// K=0..7 (vgpr0-3) and K=16..23 (vgpr4-7); lanes 16-31 hold same M,
// K=8..15 and K=24..31.  Row-major K-contiguous memory -> two 16B loads.
__device__ __forceinline__ v16h ldA_rowmajor(const _Float16* rowp, int k0, int lane) {
  const _Float16* p = rowp + k0 + ((lane & 16) ? 8 : 0);
  union { f32x4 f[2]; v16h h; } u;
  u.f[0] = *(const f32x4*)(p);
  u.f[1] = *(const f32x4*)(p + 16);
  return u.h;
}
// B-matrix 32x16 f16 (KxN): lanes 0-15 hold col N=lane, K=0..7 and K=16..23;
// lanes 16-31 hold col N=lane-16, K=8..15 and K=24..31.  Two 16B loads.
__device__ __forceinline__ v16h ldB_rowmajor(const _Float16* rowp, int k0, int lane) {
  const _Float16* p = rowp + k0 + ((lane & 16) ? 8 : 0);
  union { f32x4 f[2]; v16h h; } u;
  u.f[0] = *(const f32x4*)(p);
  u.f[1] = *(const f32x4*)(p + 16);
  return u.h;
}
// contiguous pre-packed 32B fragment (conv weights; global or LDS pointer)
__device__ __forceinline__ v16h ld_frag32(const _Float16* p) {
  union { f32x4 f[2]; v16h h; } u;
  u.f[0] = *(const f32x4*)(p);
  u.f[1] = *(const f32x4*)(p + 8);
  return u.h;
}

// ---- small utility kernels ------------------------------------------------
__global__ void cvt_f32_to_f16(const float* __restrict__ s, _Float16* __restrict__ d, size_t n) {
  size_t i = (size_t)blockIdx.x * blockDim.x + threadIdx.x;
  if (i < n) d[i] = (_Float16)s[i];
}

// f32 [dout][din] -> f16 [doutp][dinp] with zero padding (row-major)
__global__ void cvt_pad_w16(const float* __restrict__ src, _Float16* __restrict__ dst,
                            int dout, int din, int doutp, int dinp) {
  int i = blockIdx.x * blockDim.x + threadIdx.x;
  if (i >= doutp * dinp) return;
  int r = i / dinp, c = i - r * dinp;
  float v = (r < dout && c < din) ? src[r * din + c] : 0.0f;
  dst[i] = (_Float16)v;
}

__global__ void pad_bias(const float* __restrict__ src, float* __restrict__ dst, int n, int np) {
  int i = blockIdx.x * blockDim.x + threadIdx.x;
  if (i < np) dst[i] = (i < n) ? src[i] : 0.0f;
}

// fold conv bias + BN (eval mode) into per-channel scale/shift
__global__ void bn_fold(const float* __restrict__ cb, const float* __restrict__ g,
                        const float* __restrict__ b, const float* __restrict__ m,
                        const float* __restrict__ v, float* __restrict__ scale,
                        float* __restrict__ shift, int C) {
  int c = blockIdx.x * blockDim.x + threadIdx.x;
  if (c >= C) return;
  float s = g[c] * rsqrtf(v[c] + 1e-5f);
  scale[c] = s;
  shift[c] = (cb[c] - m[c]) * s + b[c];
}

// pack conv weights [COUT][CIN*9] f32 into per-chunk A fragments:
// wp[(mt*KC+kc)*32 + lane][16 halves] matching the 16x32 A layout.
template <int CIN, int COUT>
__global__ void pack_conv_w(const float* __restrict__ w, _Float16* __restrict__ wp) {
  constexpr int K  = CIN * 9;
  constexpr int KC = (K + 31) / 32;
  const int lane = threadIdx.x;         // 32
  const int kc = blockIdx.x, mt = blockIdx.y;
  const int m  = mt * 16 + (lane & 15);
  const int kb = (lane & 16) ? 8 : 0;
  _Float16* dst = wp + ((size_t)(mt * KC + kc) * 32 + lane) * 16;
#pragma unroll
  for (int j = 0; j < 16; ++j) {
    int k = kc * 32 + kb + ((j < 8) ? j : j + 8);
    float v = (k < K) ? w[(size_t)m * K + k] : 0.0f;
    dst[j] = (_Float16)v;
  }
}

// ---- implicit-GEMM conv + fused BN + ReLU (one wave per 16x16 tile) -------
// in:  f16 [B][CIN][H][W];   wp: packed A fragments;   out: f16 [B][COUT][H][W]
template <int CIN, int COUT, int H, int W>
__global__ __launch_bounds__(32) void conv_wmma(const _Float16* __restrict__ in,
                                                const _Float16* __restrict__ wp,
                                                const float* __restrict__ scale,
                                                const float* __restrict__ shift,
                                                _Float16* __restrict__ out) {
  constexpr int K  = CIN * 9;
  constexpr int KC = (K + 31) / 32;
  __shared__ _Float16 smem[CIN * 3 * 20];   // [CIN][3 rows][18 cols pad 20] halo strip
#if HAVE_TDM
  __shared__ __align__(16) _Float16 wsm[KC * 512];  // weight fragment panel
#endif

  const int lane = threadIdx.x;
  const int pt = blockIdx.x;               // pixel tile: (W/16) per row * H
  const int mt = blockIdx.y;               // cout tile
  const int b  = blockIdx.z;
  const int row  = pt / (W / 16);
  const int col0 = (pt % (W / 16)) * 16;
  const _Float16* inb = in + (size_t)b * CIN * H * W;
  const _Float16* wpanel = wp + (size_t)mt * KC * 512;

#if HAVE_TDM
  // Issue the Tensor Data Mover copy of this cout-tile's packed weight panel
  // (KC*1024 bytes, contiguous, as KC*128 8-byte elements) global -> LDS,
  // overlapped with the halo staging below.  D# per ISA ch.8.
  {
    unsigned lds_off = (unsigned)(uintptr_t)(&wsm[0]);   // low 32b of flat = LDS offset
    unsigned long long ga = (unsigned long long)(uintptr_t)wpanel;
    constexpr unsigned NE = (unsigned)(KC * 128);        // 8B elements
    u32x4 g0;
    g0[0] = 1u;                                          // count=1 valid descriptor
    g0[1] = lds_off;                                     // lds_addr (bytes)
    g0[2] = (unsigned)(ga & 0xFFFFFFFFu);                // global_addr[31:0]
    g0[3] = (unsigned)((ga >> 32) & 0x1FFFFFFu) | (2u << 30);  // addr[56:32], type=2
    i32x8 g1 = {};
    g1[0] = (int)(3u << 16);                             // data_size=3 (8 bytes)
    g1[1] = (int)((NE & 0xFFFFu) << 16);                 // tensor_dim0[15:0]
    g1[2] = (int)((NE >> 16) & 0xFFFFu) | (1 << 16);     // tensor_dim0[31:16], tensor_dim1=1
    g1[3] = (int)((NE & 0xFFFFu) << 16);                 // tile_dim0 = NE
    i32x4 z4 = {};
#if __clang_major__ >= 23
    i32x8 z8 = {};
    __builtin_amdgcn_tensor_load_to_lds(g0, g1, z4, z4, z8, 0);
#else
    __builtin_amdgcn_tensor_load_to_lds(g0, g1, z4, z4, 0);
#endif
  }
#endif

  // stage input halo strip into LDS (zero padded borders)
  for (int idx = lane; idx < CIN * 3 * 18; idx += 32) {
    int c  = idx / 54;
    int r3 = (idx / 18) % 3;
    int cc = idx % 18;
    int y = row + r3 - 1, x = col0 + cc - 1;
    _Float16 v = (_Float16)0.0f;
    if (y >= 0 && y < H && x >= 0 && x < W) v = inb[(size_t)c * H * W + y * W + x];
    smem[(c * 3 + r3) * 20 + cc] = v;
  }
#if HAVE_TDM
  __builtin_amdgcn_s_wait_tensorcnt(0);   // weight panel resident in LDS
#endif
  __syncthreads();

  const int n = lane & 15;                 // pixel within tile / B-col
  const int kb = (lane & 16) ? 8 : 0;      // B-fragment K base for this lane
  v8f acc = {};
  for (int kc = 0; kc < KC; ++kc) {
#if HAVE_TDM
    const _Float16* ap = wsm + ((size_t)kc * 32 + lane) * 16;      // ds_load_b128 x2
#else
    const _Float16* ap = wpanel + ((size_t)kc * 32 + lane) * 16;
    __builtin_prefetch(ap + 512, 0, 1);    // global_prefetch_b8: next chunk's fragments
#endif
    v16h a = ld_frag32(ap);
    union { _Float16 e[16]; v16h v; } bu;
#pragma unroll
    for (int j = 0; j < 16; ++j) {
      int k = kc * 32 + kb + ((j < 8) ? j : j + 8);
      _Float16 val = (_Float16)0.0f;
      if (k < K) {
        int cin = k / 9;
        int rem = k - cin * 9;
        int rr  = rem / 3;
        int cc  = rem - rr * 3;
        val = smem[(cin * 3 + rr) * 20 + n + cc];
      }
      bu.e[j] = val;
    }
    acc = __builtin_amdgcn_wmma_f32_16x16x32_f16(false, a, false, bu.v,
                                                 (short)0, acc, false, false);
  }
  // epilogue: BN fold + ReLU, f16 store.  D layout: vgpr i -> M=i (+8 lanes>=16)
#pragma unroll
  for (int i = 0; i < 8; ++i) {
    int co = mt * 16 + i + ((lane & 16) ? 8 : 0);
    float v = acc[i] * scale[co] + shift[co];
    v = fmaxf(v, 0.0f);
    out[((size_t)b * COUT + co) * (H * W) + row * W + col0 + n] = (_Float16)v;
  }
}

// ---- 2x2 maxpool (f16) ----------------------------------------------------
__global__ void maxpool2(const _Float16* __restrict__ in, _Float16* __restrict__ out,
                         int C, int H, int W) {
  size_t ntot = (size_t)BATCH * C * (H / 2) * (W / 2);
  size_t i = (size_t)blockIdx.x * blockDim.x + threadIdx.x;
  if (i >= ntot) return;
  int w2 = W / 2;
  int x = (int)(i % w2);
  size_t r = i / w2;
  int y = (int)(r % (H / 2));
  size_t bc = r / (H / 2);
  const _Float16* p = in + (bc * H + (size_t)y * 2) * W + x * 2;
  float m0 = fmaxf((float)p[0], (float)p[1]);
  float m1 = fmaxf((float)p[W], (float)p[W + 1]);
  out[i] = (_Float16)fmaxf(m0, m1);
}

// ---- adaptive avg pool 32x32 -> 4x4, write f16 features into concat buf ---
__global__ void avgpool_feat(const _Float16* __restrict__ in, _Float16* __restrict__ hdst) {
  int i = blockIdx.x * blockDim.x + threadIdx.x;       // 64*2048
  if (i >= BATCH * 2048) return;
  int b = i >> 11, f = i & 2047;
  int c = f >> 4, oy = (f >> 2) & 3, ox = f & 3;
  const _Float16* p = in + (((size_t)b * 128 + c) * 32 + oy * 8) * 32 + ox * 8;
  float s = 0.0f;
  for (int y = 0; y < 8; ++y)
#pragma unroll
    for (int x = 0; x < 8; ++x) s += (float)p[y * 32 + x];
  hdst[(size_t)b * 2176 + f] = (_Float16)(s * (1.0f / 64.0f));
}

// ---- generic WMMA GEMM: C[64][N] = A[64][K](f16,lda) * W[N][K](f16,ldb)^T + bias
// act: 0 none, 1 relu, 2 tanh.  One wave per 16x16 tile; grid=(N/16, 4).
__global__ __launch_bounds__(32) void gemm_wmma(const _Float16* __restrict__ A, int lda,
                                                const _Float16* __restrict__ Wt, int ldb,
                                                const float* __restrict__ bias,
                                                float* __restrict__ C, int ldc,
                                                int K, int act) {
  const int lane = threadIdx.x;
  const int nt = blockIdx.x, mt = blockIdx.y;
  const int r = lane & 15;
  const _Float16* arow = A  + (size_t)(mt * 16 + r) * lda;
  const _Float16* brow = Wt + (size_t)(nt * 16 + r) * ldb;
  v8f acc = {};
  for (int k0 = 0; k0 < K; k0 += 32) {
    __builtin_prefetch(brow + k0 + 256, 0, 1);
    v16h a = ldA_rowmajor(arow, k0, lane);
    v16h b = ldB_rowmajor(brow, k0, lane);
    acc = __builtin_amdgcn_wmma_f32_16x16x32_f16(false, a, false, b,
                                                 (short)0, acc, false, false);
  }
#pragma unroll
  for (int i = 0; i < 8; ++i) {
    int m = mt * 16 + i + ((lane & 16) ? 8 : 0);
    int n = nt * 16 + r;
    float v = acc[i] + bias[n];
    if (act == 1) v = fmaxf(v, 0.0f);
    else if (act == 2) v = tanhf(v);
    C[(size_t)m * ldc + n] = v;
  }
}

// ---- LayerNorm (+optional ReLU) over last dim, one block per row ----------
__global__ void ln_act(const float* __restrict__ x, const float* __restrict__ g,
                       const float* __restrict__ b, _Float16* __restrict__ y16,
                       int d, int ld16, int relu) {
  __shared__ float s1[256], s2[256];
  int row = blockIdx.x, tid = threadIdx.x;
  const float* xr = x + (size_t)row * d;
  float a = 0.0f, a2 = 0.0f;
  for (int i = tid; i < d; i += 256) { float v = xr[i]; a += v; a2 += v * v; }
  s1[tid] = a; s2[tid] = a2;
  __syncthreads();
  for (int off = 128; off > 0; off >>= 1) {
    if (tid < off) { s1[tid] += s1[tid + off]; s2[tid] += s2[tid + off]; }
    __syncthreads();
  }
  float mean = s1[0] / d;
  float var  = s2[0] / d - mean * mean;
  float rstd = rsqrtf(var + 1e-5f);
  for (int i = tid; i < d; i += 256) {
    float v = (xr[i] - mean) * rstd * g[i] + b[i];
    if (relu) v = fmaxf(v, 0.0f);
    y16[(size_t)row * ld16 + i] = (_Float16)v;
  }
}

// ---- 4-qubit statevector circuits, one thread per batch element -----------
__device__ __forceinline__ void ry_g(float (&re)[16], float (&im)[16], int q, float t) {
  float c = __cosf(t * 0.5f), s = __sinf(t * 0.5f);
  int m = 1 << (3 - q);
#pragma unroll
  for (int i = 0; i < 16; ++i)
    if (!(i & m)) {
      int j = i | m;
      float r0 = re[i], i0 = im[i], r1 = re[j], i1 = im[j];
      re[i] = c * r0 - s * r1; im[i] = c * i0 - s * i1;
      re[j] = s * r0 + c * r1; im[j] = s * i0 + c * i1;
    }
}
__device__ __forceinline__ void rz_g(float (&re)[16], float (&im)[16], int q, float t) {
  float c = __cosf(t * 0.5f), s = __sinf(t * 0.5f);
  int m = 1 << (3 - q);
#pragma unroll
  for (int i = 0; i < 16; ++i) {
    float r = re[i], ii = im[i];
    if (!(i & m)) { re[i] = c * r + s * ii; im[i] = c * ii - s * r; }   // * e^{-it/2}
    else          { re[i] = c * r - s * ii; im[i] = c * ii + s * r; }  // * e^{+it/2}
  }
}
__device__ __forceinline__ void rx_g(float (&re)[16], float (&im)[16], int q, float t) {
  float c = __cosf(t * 0.5f), s = __sinf(t * 0.5f);
  int m = 1 << (3 - q);
#pragma unroll
  for (int i = 0; i < 16; ++i)
    if (!(i & m)) {
      int j = i | m;
      float r0 = re[i], i0 = im[i], r1 = re[j], i1 = im[j];
      re[i] = c * r0 + s * i1; im[i] = c * i0 - s * r1;   // c*a0 - i s*a1
      re[j] = c * r1 + s * i0; im[j] = c * i1 - s * r0;   // -i s*a0 + c*a1
    }
}
__device__ __forceinline__ void cnot_g(float (&re)[16], float (&im)[16], int c, int t) {
  int mc = 1 << (3 - c), mt = 1 << (3 - t);
#pragma unroll
  for (int i = 0; i < 16; ++i)
    if ((i & mc) && !(i & mt)) {
      int j = i | mt;
      float tr = re[i]; re[i] = re[j]; re[j] = tr;
      float ti = im[i]; im[i] = im[j]; im[j] = ti;
    }
}

__global__ void quantum_kernel(const float* __restrict__ q1,   // [64][16]
                               const float* __restrict__ q2,   // [64][16] (4 valid)
                               const float* __restrict__ w1,   // [2][4][2]
                               const float* __restrict__ w2,   // [2][4][2]
                               _Float16* __restrict__ comb_pad) { // [64][32]
  int t = threadIdx.x;
  if (t >= BATCH) return;
  float re[16], im[16], out[10];
  // ---- circuit 1: amplitude embedding, RY/RZ layers + CNOT chain ----
  {
    float s = 1e-20f;
    for (int i = 0; i < 16; ++i) { float v = q1[t * 16 + i]; s += v * v; }
    float inv = rsqrtf(s);
    for (int i = 0; i < 16; ++i) { re[i] = q1[t * 16 + i] * inv; im[i] = 0.0f; }
    for (int l = 0; l < 2; ++l) {
      for (int q = 0; q < 4; ++q) {
        ry_g(re, im, q, w1[(l * 4 + q) * 2 + 0]);
        rz_g(re, im, q, w1[(l * 4 + q) * 2 + 1]);
      }
      for (int q = 0; q < 3; ++q) cnot_g(re, im, q, q + 1);
    }
    for (int q = 0; q < 4; ++q) {
      int m = 1 << (3 - q);
      float z = 0.0f;
      for (int i = 0; i < 16; ++i) {
        float p = re[i] * re[i] + im[i] * im[i];
        z += (i & m) ? -p : p;
      }
      out[q] = z;
    }
  }
  // ---- circuit 2: angle encoding, RX/RY layers + ring CNOT ----
  {
    for (int i = 0; i < 16; ++i) { re[i] = 0.0f; im[i] = 0.0f; }
    re[0] = 1.0f;
    for (int q = 0; q < 4; ++q) ry_g(re, im, q, q2[t * 16 + q] * 3.14159265358979f);
    for (int l = 0; l < 2; ++l) {
      for (int q = 0; q < 4; ++q) {
        rx_g(re, im, q, w2[(l * 4 + q) * 2 + 0]);
        ry_g(re, im, q, w2[(l * 4 + q) * 2 + 1]);
      }
      for (int q = 0; q < 4; ++q) cnot_g(re, im, q, (q + 1) & 3);
    }
    for (int q = 0; q < 4; ++q) {
      int m = 1 << (3 - q);
      float z = 0.0f;
      for (int i = 0; i < 16; ++i) {
        float p = re[i] * re[i] + im[i] * im[i];
        z += (i & m) ? -p : p;
      }
      out[4 + q] = z;
    }
    for (int q = 0; q < 2; ++q) {
      int m = 1 << (3 - q);
      float x = 0.0f;
      for (int i = 0; i < 16; ++i) x += re[i] * re[i ^ m] + im[i] * im[i ^ m];
      out[8 + q] = x;
    }
  }
  for (int j = 0; j < 32; ++j)
    comb_pad[t * 32 + j] = (_Float16)((j < 10) ? out[j] : 0.0f);
}

// ---- final 128 -> 2 classifier head (tiny; scalar) ------------------------
__global__ void final_lin(const float* __restrict__ x, const float* __restrict__ w,
                          const float* __restrict__ b, float* __restrict__ out) {
  int i = threadIdx.x;
  if (i >= BATCH * 2) return;
  int r = i >> 1, n = i & 1;
  float s = b[n];
  for (int k = 0; k < 128; ++k) s += x[r * 128 + k] * w[n * 128 + k];
  out[i] = s;
}

// ===========================================================================
extern "C" void kernel_launch(void* const* d_in, const int* in_sizes, int n_in,
                              void* d_out, int out_size, void* d_ws, size_t ws_size,
                              hipStream_t stream) {
  (void)in_sizes; (void)n_in; (void)out_size; (void)ws_size;
  auto F = [&](int i) { return (const float*)d_in[i]; };
  // input index map (setup_inputs insertion order)
  const float* x = F(0);
  // conv i: w=1+6i b=2+6i bn_g=3+6i bn_b=4+6i bn_m=5+6i bn_v=6+6i
  const float* qp1_l0_w = F(37); const float* qp1_l0_b = F(38);
  const float* qp1_ln_g = F(39); const float* qp1_ln_b = F(40);
  const float* qp1_l1_w = F(41); const float* qp1_l1_b = F(42);
  const float* qp2_l0_w = F(43); const float* qp2_l0_b = F(44);
  const float* qp2_ln_g = F(45); const float* qp2_ln_b = F(46);
  const float* qp2_l1_w = F(47); const float* qp2_l1_b = F(48);
  const float* qw1 = F(49);      const float* qw2 = F(50);
  const float* fus_w = F(51);    const float* fus_b = F(52);
  const float* fus_g = F(53);    const float* fus_bb = F(54);
  const float* c0_w = F(55); const float* c0_b = F(56);
  const float* c0_g = F(57); const float* c0_bb = F(58);
  const float* c1_w = F(59); const float* c1_b = F(60);
  const float* c1_g = F(61); const float* c1_bb = F(62);
  const float* c2_w = F(63); const float* c2_b = F(64);
  const float* c3_w = F(65); const float* c3_b = F(66);

  // ---- workspace bump allocator ----
  char* base = (char*)d_ws;
  size_t off = 0;
  auto alloc = [&](size_t bytes) -> void* {
    void* p = base + off;
    off += (bytes + 255) & ~(size_t)255;
    return p;
  };
  _Float16* xin16 = (_Float16*)alloc((size_t)BATCH * 128 * 128 * 2);
  _Float16* bufA  = (_Float16*)alloc((size_t)BATCH * 32 * 128 * 128 * 2);
  _Float16* bufB  = (_Float16*)alloc((size_t)BATCH * 32 * 128 * 128 * 2);
  // packed conv weights
  _Float16* cw0 = (_Float16*)alloc(2  * 1  * 512 * 2);
  _Float16* cw1 = (_Float16*)alloc(2  * 9  * 512 * 2);
  _Float16* cw2 = (_Float16*)alloc(4  * 9  * 512 * 2);
  _Float16* cw3 = (_Float16*)alloc(4  * 18 * 512 * 2);
  _Float16* cw4 = (_Float16*)alloc(8  * 18 * 512 * 2);
  _Float16* cw5 = (_Float16*)alloc(8  * 36 * 512 * 2);
  float* bnscale[6]; float* bnshift[6];
  const int couts[6] = {32, 32, 64, 64, 128, 128};
  for (int i = 0; i < 6; ++i) {
    bnscale[i] = (float*)alloc(couts[i] * 4);
    bnshift[i] = (float*)alloc(couts[i] * 4);
  }
  // concat buffer h: [64][2176] f16 (fused @0..127, features @128..2175)
  _Float16* h = (_Float16*)alloc((size_t)BATCH * 2176 * 2);
  // linear weights f16 (padded)
  _Float16* w_qp1l0 = (_Float16*)alloc((size_t)512 * 2048 * 2);
  _Float16* w_qp1l1 = (_Float16*)alloc((size_t)16  * 512  * 2);
  _Float16* w_qp2l0 = (_Float16*)alloc((size_t)128 * 2048 * 2);
  _Float16* w_qp2l1 = (_Float16*)alloc((size_t)16  * 128  * 2);  // dout 4->16 pad
  _Float16* w_fus   = (_Float16*)alloc((size_t)128 * 32   * 2);  // din 10->32 pad
  _Float16* w_c0    = (_Float16*)alloc((size_t)512 * 2176 * 2);
  _Float16* w_c1    = (_Float16*)alloc((size_t)256 * 512  * 2);
  _Float16* w_c2    = (_Float16*)alloc((size_t)128 * 256  * 2);
  float* b_qp2l1p = (float*)alloc(16 * 4);
  // f32 scratch
  float* t_qp1 = (float*)alloc((size_t)BATCH * 512 * 4);
  float* t_qp2 = (float*)alloc((size_t)BATCH * 128 * 4);
  float* q1    = (float*)alloc((size_t)BATCH * 16 * 4);
  float* q2    = (float*)alloc((size_t)BATCH * 16 * 4);
  float* fusl  = (float*)alloc((size_t)BATCH * 128 * 4);
  float* cls0  = (float*)alloc((size_t)BATCH * 512 * 4);
  float* cls1  = (float*)alloc((size_t)BATCH * 256 * 4);
  float* cls2  = (float*)alloc((size_t)BATCH * 128 * 4);
  // f16 scratch
  _Float16* qp1h16 = (_Float16*)alloc((size_t)BATCH * 512 * 2);
  _Float16* qp2h16 = (_Float16*)alloc((size_t)BATCH * 128 * 2);
  _Float16* comb16 = (_Float16*)alloc((size_t)BATCH * 32 * 2);
  _Float16* c016   = (_Float16*)alloc((size_t)BATCH * 512 * 2);
  _Float16* c116   = (_Float16*)alloc((size_t)BATCH * 256 * 2);

  // ---- prep: conversions, BN folds, weight packing ----
  {
    size_t n = (size_t)BATCH * 128 * 128;
    cvt_f32_to_f16<<<(unsigned)((n + 255) / 256), 256, 0, stream>>>(x, xin16, n);
  }
  for (int i = 0; i < 6; ++i) {
    bn_fold<<<(couts[i] + 127) / 128, 128, 0, stream>>>(
        F(2 + 6 * i), F(3 + 6 * i), F(4 + 6 * i), F(5 + 6 * i), F(6 + 6 * i),
        bnscale[i], bnshift[i], couts[i]);
  }
  pack_conv_w<1,   32><<<dim3(1,  2), 32, 0, stream>>>(F(1),  cw0);
  pack_conv_w<32,  32><<<dim3(9,  2), 32, 0, stream>>>(F(7),  cw1);
  pack_conv_w<32,  64><<<dim3(9,  4), 32, 0, stream>>>(F(13), cw2);
  pack_conv_w<64,  64><<<dim3(18, 4), 32, 0, stream>>>(F(19), cw3);
  pack_conv_w<64, 128><<<dim3(18, 8), 32, 0, stream>>>(F(25), cw4);
  pack_conv_w<128,128><<<dim3(36, 8), 32, 0, stream>>>(F(31), cw5);
  auto padw = [&](const float* s, _Float16* d, int dout, int din, int doutp, int dinp) {
    int n = doutp * dinp;
    cvt_pad_w16<<<(n + 255) / 256, 256, 0, stream>>>(s, d, dout, din, doutp, dinp);
  };
  padw(qp1_l0_w, w_qp1l0, 512, 2048, 512, 2048);
  padw(qp1_l1_w, w_qp1l1, 16, 512, 16, 512);
  padw(qp2_l0_w, w_qp2l0, 128, 2048, 128, 2048);
  padw(qp2_l1_w, w_qp2l1, 4, 128, 16, 128);
  padw(fus_w,    w_fus,   128, 10, 128, 32);
  padw(c0_w,     w_c0,    512, 2176, 512, 2176);
  padw(c1_w,     w_c1,    256, 512, 256, 512);
  padw(c2_w,     w_c2,    128, 256, 128, 256);
  pad_bias<<<1, 32, 0, stream>>>(qp2_l1_b, b_qp2l1p, 4, 16);

  // ---- conv backbone (implicit-GEMM WMMA, fused BN+ReLU) ----
  conv_wmma<1, 32, 128, 128><<<dim3(8 * 128, 2, BATCH), 32, 0, stream>>>(
      xin16, cw0, bnscale[0], bnshift[0], bufA);
  conv_wmma<32, 32, 128, 128><<<dim3(8 * 128, 2, BATCH), 32, 0, stream>>>(
      bufA, cw1, bnscale[1], bnshift[1], bufB);
  {
    size_t n = (size_t)BATCH * 32 * 64 * 64;
    maxpool2<<<(unsigned)((n + 255) / 256), 256, 0, stream>>>(bufB, bufA, 32, 128, 128);
  }
  conv_wmma<32, 64, 64, 64><<<dim3(4 * 64, 4, BATCH), 32, 0, stream>>>(
      bufA, cw2, bnscale[2], bnshift[2], bufB);
  conv_wmma<64, 64, 64, 64><<<dim3(4 * 64, 4, BATCH), 32, 0, stream>>>(
      bufB, cw3, bnscale[3], bnshift[3], bufA);
  {
    size_t n = (size_t)BATCH * 64 * 32 * 32;
    maxpool2<<<(unsigned)((n + 255) / 256), 256, 0, stream>>>(bufA, bufB, 64, 64, 64);
  }
  conv_wmma<64, 128, 32, 32><<<dim3(2 * 32, 8, BATCH), 32, 0, stream>>>(
      bufB, cw4, bnscale[4], bnshift[4], bufA);
  conv_wmma<128, 128, 32, 32><<<dim3(2 * 32, 8, BATCH), 32, 0, stream>>>(
      bufA, cw5, bnscale[5], bnshift[5], bufB);
  avgpool_feat<<<(BATCH * 2048 + 255) / 256, 256, 0, stream>>>(bufB, h + 128);

  const _Float16* feat = h + 128;  // [64][2048] with lda=2176

  // ---- quantum projections ----
  gemm_wmma<<<dim3(32, 4), 32, 0, stream>>>(feat, 2176, w_qp1l0, 2048, qp1_l0_b,
                                            t_qp1, 512, 2048, 0);
  ln_act<<<BATCH, 256, 0, stream>>>(t_qp1, qp1_ln_g, qp1_ln_b, qp1h16, 512, 512, 1);
  gemm_wmma<<<dim3(1, 4), 32, 0, stream>>>(qp1h16, 512, w_qp1l1, 512, qp1_l1_b,
                                           q1, 16, 512, 2);
  gemm_wmma<<<dim3(8, 4), 32, 0, stream>>>(feat, 2176, w_qp2l0, 2048, qp2_l0_b,
                                           t_qp2, 128, 2048, 0);
  ln_act<<<BATCH, 256, 0, stream>>>(t_qp2, qp2_ln_g, qp2_ln_b, qp2h16, 128, 128, 1);
  gemm_wmma<<<dim3(1, 4), 32, 0, stream>>>(qp2h16, 128, w_qp2l1, 128, b_qp2l1p,
                                           q2, 16, 128, 2);

  // ---- quantum circuits ----
  quantum_kernel<<<1, 64, 0, stream>>>(q1, q2, qw1, qw2, comb16);

  // ---- fusion ----
  gemm_wmma<<<dim3(8, 4), 32, 0, stream>>>(comb16, 32, w_fus, 32, fus_b,
                                           fusl, 128, 32, 0);
  ln_act<<<BATCH, 256, 0, stream>>>(fusl, fus_g, fus_bb, h /*cols 0..127*/, 128, 2176, 1);

  // ---- classifier ----
  gemm_wmma<<<dim3(32, 4), 32, 0, stream>>>(h, 2176, w_c0, 2176, c0_b,
                                            cls0, 512, 2176, 0);
  ln_act<<<BATCH, 256, 0, stream>>>(cls0, c0_g, c0_bb, c016, 512, 512, 1);
  gemm_wmma<<<dim3(16, 4), 32, 0, stream>>>(c016, 512, w_c1, 512, c1_b,
                                            cls1, 256, 512, 0);
  ln_act<<<BATCH, 256, 0, stream>>>(cls1, c1_g, c1_bb, c116, 256, 256, 1);
  gemm_wmma<<<dim3(8, 4), 32, 0, stream>>>(c116, 256, w_c2, 256, c2_b,
                                           cls2, 128, 256, 1);
  final_lin<<<1, 128, 0, stream>>>(cls2, c3_w, c3_b, (float*)d_out);
}
